// ExpertChoiceTokenSparseMoE_9517647528390
// MI455X (gfx1250) — matrix operand order repaired and checked by reading, not compiled, not run on hardware
//
#include <hip/hip_runtime.h>
#include <hip/hip_bf16.h>

#define E_ 8
#define D_ 1024
#define F_ 4096
#define N_ 32768
#define K_ 8192
#define BM 32
#define FC 128
#define XLD32 520           // lds stride (dwords) for x tile  (512 + 8 pad)
#define HLD (FC + 8)        // lds stride (ushorts) for h tile
#define WLD (FC + 4)        // lds stride (dwords) for staged w1 slab (pad -> halves on disjoint banks)

typedef __attribute__((ext_vector_type(16))) __bf16 v16bf;
typedef __attribute__((ext_vector_type(8)))  float  v8f;
typedef int v4i_ __attribute__((vector_size(16)));   // builtin's b128 payload type

#if defined(__has_builtin)
#if __has_builtin(__builtin_amdgcn_global_load_async_to_lds_b128)
#define ASYNC_LDS 1
#endif
#endif

// ---------- small helpers ----------
__device__ __forceinline__ unsigned f32_to_bf16u(float f) {
  unsigned u = __float_as_uint(f);
  u += 0x7FFFu + ((u >> 16) & 1u);          // round-to-nearest-even
  return u >> 16;
}
__device__ __forceinline__ unsigned pack_bf16(float lo, float hi) {
  return f32_to_bf16u(lo) | (f32_to_bf16u(hi) << 16);
}
__device__ __forceinline__ unsigned fkey(float f) {  // order-preserving key
  unsigned u = __float_as_uint(f);
  return (u & 0x80000000u) ? ~u : (u | 0x80000000u);
}
__device__ __forceinline__ float funkey(unsigned k) {
  return __uint_as_float((k & 0x80000000u) ? (k & 0x7FFFFFFFu) : ~k);
}

// 16B global -> LDS copy, async on gfx1250 (ASYNCcnt), sync fallback otherwise.
__device__ __forceinline__ void async_copy_b128(const unsigned* __restrict__ g, unsigned* l) {
#ifdef ASYNC_LDS
  __builtin_amdgcn_global_load_async_to_lds_b128((v4i_*)g, (v4i_*)l, 0, 0);
#else
  *(uint4*)l = *(const uint4*)g;
#endif
}
__device__ __forceinline__ void wait_async_keep8() {
#if defined(__has_builtin) && __has_builtin(__builtin_amdgcn_s_wait_asynccnt)
  __builtin_amdgcn_s_wait_asynccnt(8);
#else
  asm volatile("s_wait_asynccnt 0x8" ::: "memory");
#endif
}
__device__ __forceinline__ void wait_async_all() {
#if defined(__has_builtin) && __has_builtin(__builtin_amdgcn_s_wait_asynccnt)
  __builtin_amdgcn_s_wait_asynccnt(0);
#else
  asm volatile("s_wait_asynccnt 0x0" ::: "memory");
#endif
}

// A fragment (16x32 bf16) from dword-packed LDS (pairs (2kp,2kp+1) in one u32).
__device__ __forceinline__ v16bf load_a_lds32(const unsigned* p, int rowbase, int kb, int ldstr32) {
  int lane = threadIdx.x & 31;
  int row  = rowbase + (lane & 15);
  int kph  = (lane >> 4) << 2;
  const unsigned* q = p + row * ldstr32 + (kb >> 1);
  union { unsigned u[8]; v16bf v; } r;
#pragma unroll
  for (int i = 0; i < 8; ++i) r.u[i] = q[(i & 3) + kph + ((i >> 2) << 3)];
  return r.v;
}

// A fragment from 16-bit LDS (h tile); k offsets always even -> aligned dword reads.
__device__ __forceinline__ v16bf load_a_lds16(const unsigned short* p, int rowbase, int kb, int ldstr) {
  int lane  = threadIdx.x & 31;
  int row   = rowbase + (lane & 15);
  int khalf = (lane >> 4) << 3;
  union { unsigned u[8]; v16bf v; } r;
#pragma unroll
  for (int i = 0; i < 8; ++i) {
    int k = kb + ((i & 3) << 1) + khalf + ((i >> 2) << 4);
    r.u[i] = *(const unsigned*)(p + row * ldstr + k);
  }
  return r.v;
}

// B fragment (32x16) from pre-packed bf16 global weights: wp[kp*ld + n] = (k=2kp,2kp+1).
__device__ __forceinline__ v16bf load_b_pk(const unsigned* __restrict__ mp, int ld) {
  int lane = threadIdx.x & 31;
  int n    = lane & 15;
  int kph  = (lane >> 4) << 2;
  union { unsigned u[8]; v16bf v; } r;
#pragma unroll
  for (int i = 0; i < 8; ++i) r.u[i] = mp[(size_t)((i & 3) + kph + ((i >> 2) << 3)) * ld + n];
  return r.v;
}

// B fragment from LDS-staged w1 slab (64 kp-rows x FC cols, stride WLD dwords).
__device__ __forceinline__ v16bf load_b_stage(const unsigned* buf, int ks, int colbase) {
  int lane = threadIdx.x & 31;
  int n    = lane & 15;
  int kph  = (lane >> 4) << 2;
  const unsigned* q = buf + (ks >> 1) * WLD + colbase + n;
  union { unsigned u[8]; v16bf v; } r;
#pragma unroll
  for (int i = 0; i < 8; ++i) r.u[i] = q[((i & 3) + kph + ((i >> 2) << 3)) * WLD];
  return r.v;
}

// Fill one staged slab: w1 rows [kb, kb+128) (64 kp), cols [fc, fc+FC) -> 8 b128 per thread.
__device__ __forceinline__ void stage_w1_slab(const unsigned* __restrict__ w1e, unsigned* buf,
                                              int kb, int fc, int tid) {
#pragma unroll
  for (int s = 0; s < 8; ++s) {
    int j16 = tid + s * 256;                 // 0..2047, unit = 4 dwords
    int kp  = j16 >> 5;                      // 0..63
    int n   = (j16 & 31) << 2;               // 0..124
    async_copy_b128(w1e + (size_t)((kb >> 1) + kp) * F_ + fc + n, buf + kp * WLD + n);
  }
}

// Store a 16x16 f32 accumulator tile as relu(acc+bias) -> bf16 LDS.
__device__ __forceinline__ void store_h_tile(unsigned short* lds, int ldstr, int rowbase,
                                             int colbase, v8f acc, const float* __restrict__ bias) {
  int lane = threadIdx.x & 31;
  int n    = lane & 15;
  int rofs = rowbase + ((lane >> 4) << 3);
  float b  = bias[n];
#pragma unroll
  for (int i = 0; i < 8; ++i) {
    float h = acc[i] + b;
    h = h > 0.f ? h : 0.f;
    lds[(rofs + i) * ldstr + colbase + n] = (unsigned short)f32_to_bf16u(h);
  }
}

// ---------- kernel 0: pack f32 weights -> K-pair-interleaved bf16 dwords ----------
__global__ void pack_w_kernel(const float* __restrict__ w, unsigned* __restrict__ wp, int ld) {
  size_t i  = (size_t)blockIdx.x * blockDim.x + threadIdx.x;
  size_t kp = i / (unsigned)ld;
  size_t n  = i % (unsigned)ld;
  const float* s = w + 2 * kp * (size_t)ld + n;
  wp[i] = pack_bf16(s[0], s[ld]);
}

// ---------- kernel 1: zero scratch counters ----------
__global__ void init_kernel(int* cnt, float* sumexp) {
  int t = threadIdx.x;
  if (t < E_) { cnt[t] = 0; sumexp[t] = 0.f; }
}

// ---------- kernel 2: router -> noisy logits [E][N] ----------
__global__ void router_kernel(const float* __restrict__ x, const float* __restrict__ noise,
                              const float* __restrict__ wr, const float* __restrict__ br,
                              const float* __restrict__ wn, const float* __restrict__ bn,
                              float* __restrict__ noisy) {
  int n = blockIdx.x * blockDim.x + threadIdx.x;
  if (n >= N_) return;
  const float* xr = x + (size_t)n * D_;
  float lr[E_] = {}, ln[E_] = {};
  for (int d = 0; d < D_; ++d) {
    float xv = xr[d];
#pragma unroll
    for (int j = 0; j < E_; ++j) {
      lr[j] = fmaf(xv, wr[d * E_ + j], lr[j]);
      ln[j] = fmaf(xv, wn[d * E_ + j], ln[j]);
    }
  }
#pragma unroll
  for (int j = 0; j < E_; ++j) {
    float nl = ln[j] + bn[j];
    float sp = nl > 20.f ? nl : log1pf(expf(nl));
    noisy[(size_t)j * N_ + n] = lr[j] + br[j] + noise[(size_t)j * N_ + n] * sp;
  }
}

// ---------- kernel 3: per-expert 2-pass radix threshold (top-16-bit key prefix) ----------
__global__ void threshold_kernel(const float* __restrict__ noisy,
                                 unsigned* __restrict__ thresh, float* __restrict__ vmax) {
  int e = blockIdx.x;
  const float* row = noisy + (size_t)e * N_;
  __shared__ unsigned hist[256];
  __shared__ unsigned maxk_sh, b1_sh, above_sh;
  int tid = threadIdx.x;
  if (tid < 256) hist[tid] = 0;
  if (tid == 0) maxk_sh = 0;
  __syncthreads();
  unsigned mk = 0;
  for (int i = tid; i < N_; i += blockDim.x) {
    unsigned k = fkey(row[i]);
    atomicAdd(&hist[k >> 24], 1u);
    mk = mk > k ? mk : k;
  }
  atomicMax(&maxk_sh, mk);
  __syncthreads();
  if (tid == 0) {
    unsigned cum = 0; int b = 255;
    for (; b > 0; --b) { unsigned c = hist[b]; if (cum + c >= (unsigned)K_) break; cum += c; }
    b1_sh = (unsigned)b; above_sh = cum;
  }
  __syncthreads();
  unsigned b1 = b1_sh;
  if (tid < 256) hist[tid] = 0;
  __syncthreads();
  for (int i = tid; i < N_; i += blockDim.x) {
    unsigned k = fkey(row[i]);
    if ((k >> 24) == b1) atomicAdd(&hist[(k >> 16) & 255u], 1u);
  }
  __syncthreads();
  if (tid == 0) {
    unsigned need = (unsigned)K_ - above_sh;
    unsigned cum = 0; int b = 255;
    for (; b > 0; --b) { unsigned c = hist[b]; if (cum + c >= need) break; cum += c; }
    thresh[e] = (b1 << 8) | (unsigned)b;
    vmax[e]   = funkey(maxk_sh);
  }
}

// ---------- kernels 4/5: compact selected indices + online sum-exp ----------
__global__ void select_above_kernel(const float* __restrict__ noisy, const unsigned* __restrict__ thresh,
                                    const float* __restrict__ vmax, int* __restrict__ cnt,
                                    float* __restrict__ sumexp, int* __restrict__ idx_out,
                                    float* __restrict__ vals) {
  int e = blockIdx.y;
  int i = blockIdx.x * blockDim.x + threadIdx.x;
  if (i >= N_) return;
  float v = noisy[(size_t)e * N_ + i];
  if ((fkey(v) >> 16) > thresh[e]) {
    int slot = atomicAdd(&cnt[e], 1);
    idx_out[(size_t)e * K_ + slot] = i;
    vals[(size_t)e * K_ + slot]    = v;
    atomicAdd(&sumexp[e], expf(v - vmax[e]));
  }
}
__global__ void select_tie_kernel(const float* __restrict__ noisy, const unsigned* __restrict__ thresh,
                                  const float* __restrict__ vmax, int* __restrict__ cnt,
                                  float* __restrict__ sumexp, int* __restrict__ idx_out,
                                  float* __restrict__ vals) {
  int e = blockIdx.y;
  int i = blockIdx.x * blockDim.x + threadIdx.x;
  if (i >= N_) return;
  float v = noisy[(size_t)e * N_ + i];
  if ((fkey(v) >> 16) == thresh[e]) {
    int slot = atomicAdd(&cnt[e], 1);
    if (slot < K_) {
      idx_out[(size_t)e * K_ + slot] = i;
      vals[(size_t)e * K_ + slot]    = v;
      atomicAdd(&sumexp[e], expf(v - vmax[e]));
    }
  }
}

// ---------- kernel 6: out = x (residual init) ----------
__global__ void residual_kernel(const float* __restrict__ x, float* __restrict__ out) {
  size_t i = (size_t)blockIdx.x * blockDim.x + threadIdx.x;
  ((float4*)out)[i] = ((const float4*)x)[i];
}

// ---------- kernel 7: fused WMMA expert MLP with gated scatter-add ----------
__global__ __launch_bounds__(256) void moe_mlp_kernel(
    const float* __restrict__ x,
    const unsigned* __restrict__ w1p, const float* __restrict__ b1,
    const unsigned* __restrict__ w2p, const float* __restrict__ b2,
    const int* __restrict__ idx, const float* __restrict__ vals,
    const float* __restrict__ vmax, const float* __restrict__ sumexp,
    float* __restrict__ out) {
  const int TPE   = K_ / BM;               // 256 token tiles per expert
  const int e     = blockIdx.x / TPE;      // expert-major grid order -> L2 reuse of weights
  const int kbase = (blockIdx.x % TPE) * BM;
  const int tid   = threadIdx.x;
  const int lane  = tid & 31;
  const int w     = tid >> 5;              // wave id 0..7

  __shared__ unsigned       lds_x[BM * XLD32];  // 32 gathered tokens, bf16 pair-packed (66.5 KB)
  __shared__ unsigned       lds_w[2][64 * WLD]; // double-buffered w1 slab (2 x 33 KB)
  __shared__ unsigned short lds_h[BM * HLD];    // 32 x FC hidden chunk, bf16 (8.7 KB)
  __shared__ int   s_tok[BM];
  __shared__ float s_gate[BM];

  if (tid < BM) {
    size_t slot = (size_t)e * K_ + kbase + tid;
    s_tok[tid]  = idx[slot];
    s_gate[tid] = expf(vals[slot] - vmax[e]) / sumexp[e];
  }
  __syncthreads();

  // gather token rows -> dword-packed bf16 LDS (float4 loads, coalesced per row)
  for (int ii = tid; ii < BM * (D_ / 4); ii += 256) {
    int r = ii >> 8, c4 = ii & 255;
    float4 xv = ((const float4*)(x + (size_t)s_tok[r] * D_))[c4];
    unsigned* q = &lds_x[r * XLD32 + c4 * 2];
    q[0] = pack_bf16(xv.x, xv.y);
    q[1] = pack_bf16(xv.z, xv.w);
  }
  __syncthreads();

  const unsigned* w1e = w1p + (size_t)e * (D_ / 2) * F_;
  const unsigned* w2e = w2p + (size_t)e * (F_ / 2) * D_;
  const float*    b1e = b1 + (size_t)e * F_;
  const float*    b2e = b2 + (size_t)e * D_;

  v8f oacc[16];                             // wave's 16x256 slice of 32x1024 output, f32
#pragma unroll
  for (int t = 0; t < 16; ++t) oacc[t] = (v8f){};

  const int hr = (w >> 2) * 16;             // GEMM1: rows 0-15 (waves 0-3) / 16-31 (4-7)
  const int hc = (w & 3) * 32;              // GEMM1: two 16-col tiles at hc, hc+16
  const int r2 = (w & 1) * 16;              // GEMM2: row tile
  const int cb = (w >> 1) * 256;            // GEMM2: 16 col tiles

  for (int fc = 0; fc < F_; fc += FC) {
    // ---- GEMM1: h[32,FC] = relu(x_tile @ w1[:, fc:fc+FC] + b1) ----
    // Async double-buffered LDS staging of w1 slabs (128 K-rows each).
    stage_w1_slab(w1e, lds_w[0], 0,   fc, tid);
    stage_w1_slab(w1e, lds_w[1], 128, fc, tid);
    v8f h0 = (v8f){}, h1 = (v8f){};
    for (int kb = 0; kb < D_; kb += 128) {
      unsigned* buf = lds_w[(kb >> 7) & 1];
      if (kb + 128 < D_) wait_async_keep8(); else wait_async_all();
      __syncthreads();
#pragma unroll
      for (int ks = 0; ks < 128; ks += 32) {
        v16bf a  = load_a_lds32(lds_x, hr, kb + ks, XLD32);
        v16bf bA = load_b_stage(buf, ks, hc);
        v16bf bB = load_b_stage(buf, ks, hc + 16);
        h0 = __builtin_amdgcn_wmma_f32_16x16x32_bf16(false, a, false, bA, (short)0, h0, false, false);
        h1 = __builtin_amdgcn_wmma_f32_16x16x32_bf16(false, a, false, bB, (short)0, h1, false, false);
      }
      __syncthreads();
      if (kb + 256 < D_) stage_w1_slab(w1e, buf, kb + 256, fc, tid);
    }
    store_h_tile(lds_h, HLD, hr, hc,      h0, b1e + fc + hc);
    store_h_tile(lds_h, HLD, hr, hc + 16, h1, b1e + fc + hc + 16);
    __syncthreads();

    // ---- GEMM2: out[32,1024] += h[32,FC] @ w2[fc:fc+FC, :] ----
    v16bf ha[4];
#pragma unroll
    for (int kk = 0; kk < 4; ++kk) ha[kk] = load_a_lds16(lds_h, r2, kk * 32, HLD);
#pragma unroll 4
    for (int ct = 0; ct < 16; ++ct) {
      if (lane == 0)
        __builtin_prefetch(w2e + (size_t)(fc >> 1) * D_ + cb + ct * 16, 0, 0);
#pragma unroll
      for (int kk = 0; kk < 4; ++kk) {
        v16bf bw = load_b_pk(w2e + (size_t)((fc + kk * 32) >> 1) * D_ + (cb + ct * 16), D_);
        oacc[ct] = __builtin_amdgcn_wmma_f32_16x16x32_bf16(false, ha[kk], false, bw,
                                                           (short)0, oacc[ct], false, false);
      }
    }
    __syncthreads();   // protect lds_h before next chunk's GEMM1 stores
  }

  // ---- epilogue: bias2, gate weight, scatter-add into residual output ----
  const int nofs  = lane & 15;
  const int rbase = r2 + ((lane >> 4) << 3);
#pragma unroll 2
  for (int ct = 0; ct < 16; ++ct) {
    int c = cb + ct * 16 + nofs;
    float bias = b2e[c];
#pragma unroll
    for (int i = 0; i < 8; ++i) {
      int row = rbase + i;
      float v = (oacc[ct][i] + bias) * s_gate[row];
      atomicAdd(&out[(size_t)s_tok[row] * D_ + c], v);
    }
  }
}

extern "C" void kernel_launch(void* const* d_in, const int* in_sizes, int n_in,
                              void* d_out, int out_size, void* d_ws, size_t ws_size,
                              hipStream_t stream) {
  const float* x     = (const float*)d_in[0];
  const float* noise = (const float*)d_in[1];
  const float* wr    = (const float*)d_in[2];
  const float* br    = (const float*)d_in[3];
  const float* wn    = (const float*)d_in[4];
  const float* bn    = (const float*)d_in[5];
  const float* w1    = (const float*)d_in[6];
  const float* b1    = (const float*)d_in[7];
  const float* w2    = (const float*)d_in[8];
  const float* b2    = (const float*)d_in[9];
  // d_in[10] = top_k (8192, baked in as K_)

  float* out     = (float*)d_out;
  int*   idx_out = (int*)(out + (size_t)N_ * D_);   // tuple output #2: idx [E,K]

  // workspace layout (~135.5 MB: 2x67 MB packed bf16 weights + 1.3 MB router scratch)
  const size_t W1P = (size_t)E_ * (D_ / 2) * F_;    // dwords
  const size_t W2P = (size_t)E_ * (F_ / 2) * D_;    // dwords
  unsigned* w1p    = (unsigned*)d_ws;
  unsigned* w2p    = w1p + W1P;
  float*    noisy  = (float*)(w2p + W2P);           // E*N
  float*    vals   = noisy + (size_t)E_ * N_;       // E*K (selected noisy values)
  float*    sumexp = vals + (size_t)E_ * K_;        // E
  float*    vmax   = sumexp + E_;                   // E
  unsigned* thresh = (unsigned*)(vmax + E_);        // E
  int*      cnt    = (int*)(thresh + E_);           // E

  hipLaunchKernelGGL(pack_w_kernel, dim3((unsigned)(W1P / 256)), dim3(256), 0, stream, w1, w1p, F_);
  hipLaunchKernelGGL(pack_w_kernel, dim3((unsigned)(W2P / 256)), dim3(256), 0, stream, w2, w2p, D_);
  hipLaunchKernelGGL(init_kernel, dim3(1), dim3(64), 0, stream, cnt, sumexp);
  hipLaunchKernelGGL(router_kernel, dim3(N_ / 256), dim3(256), 0, stream,
                     x, noise, wr, br, wn, bn, noisy);
  hipLaunchKernelGGL(threshold_kernel, dim3(E_), dim3(256), 0, stream, noisy, thresh, vmax);
  dim3 selg(N_ / 256, E_);
  hipLaunchKernelGGL(select_above_kernel, selg, dim3(256), 0, stream,
                     noisy, thresh, vmax, cnt, sumexp, idx_out, vals);
  hipLaunchKernelGGL(select_tie_kernel, selg, dim3(256), 0, stream,
                     noisy, thresh, vmax, cnt, sumexp, idx_out, vals);
  hipLaunchKernelGGL(residual_kernel, dim3((unsigned)((size_t)N_ * D_ / 4 / 256)), dim3(256), 0, stream,
                     x, out);
  hipLaunchKernelGGL(moe_mlp_kernel, dim3(E_ * (K_ / BM)), dim3(256), 0, stream,
                     x, w1p, b1, w2p, b2, idx_out, vals, vmax, sumexp, out);
}
